// Attention_6528350290404
// MI455X (gfx1250) — matrix operand order repaired and checked by reading, not compiled, not run on hardware
//
#include <hip/hip_runtime.h>
#include <hip/hip_bf16.h>
#include <math.h>

#define Bdim 128
#define Sdim 1024
#define Ddim 512
#define Hdim 512
#define SBLK 64          // s rows per workgroup in the score kernel
#define LDA  520         // LDS row stride (bf16 elems), 8-elem pad kills bank conflicts

typedef __attribute__((ext_vector_type(8)))  float  v8f;
typedef __attribute__((ext_vector_type(8)))  __bf16 v8bf;
typedef __attribute__((ext_vector_type(16))) __bf16 v16bf;

__device__ __forceinline__ unsigned short f2bf(float f) {
    unsigned u = __float_as_uint(f);
    u += 0x7FFFu + ((u >> 16) & 1u);      // round-to-nearest-even
    return (unsigned short)(u >> 16);
}

// Branch-free tanh: native v_tanh_f32 if the toolchain exposes it, else
// one v_exp_f32 + one v_rcp_f32 (saturates correctly at +/-1, no divergence).
__device__ __forceinline__ float fast_tanhf(float x) {
#if __has_builtin(__builtin_amdgcn_tanhf)
    return __builtin_amdgcn_tanhf(x);
#else
    float e = __expf(2.0f * x);
    return 1.0f - 2.0f * __builtin_amdgcn_rcpf(e + 1.0f);
#endif
}

// ---------------- K0: W_ctx fp32 -> bf16 (row-major [H][D]) ----------------
__global__ __launch_bounds__(256) void wcvt_kernel(const float* __restrict__ src,
                                                   unsigned short* __restrict__ dst,
                                                   int n) {
    int i = blockIdx.x * 256 + threadIdx.x;
    int stride = gridDim.x * 256;
    for (; i < n; i += stride) dst[i] = f2bf(src[i]);
}

// ------- K1/K5: out[b,h] = X[b,:]·W[h,:] + bias[h]  (B*H threads, fp32) -------
__global__ __launch_bounds__(256) void rowgemm_kernel(const float* __restrict__ X,
                                                      const float* __restrict__ W,
                                                      const float* __restrict__ bias,
                                                      float* __restrict__ out) {
    int gid = blockIdx.x * 256 + threadIdx.x;   // grid == B*H exactly
    int bb = gid >> 9;                          // Hdim == 512
    int h  = gid & (Hdim - 1);
    const float4* x = (const float4*)(X + (size_t)bb * Ddim);
    const float4* w = (const float4*)(W + (size_t)h  * Ddim);
    float acc = 0.f;
#pragma unroll 4
    for (int i = 0; i < Ddim / 4; ++i) {
        float4 a = x[i], bw = w[i];
        acc = fmaf(a.x, bw.x, acc); acc = fmaf(a.y, bw.y, acc);
        acc = fmaf(a.z, bw.z, acc); acc = fmaf(a.w, bw.w, acc);
    }
    out[gid] = acc + bias[h];
}

// ---------------- K2: masked attention scores via bf16 WMMA ----------------
// One WG per (b, 64-row s-block). 8 waves; wave w covers h-tiles {w, w+8, w+16, w+24}.
// Loop order: h-tile -> k -> 4 s-subtiles, so each W_ctx fragment is loaded once
// and feeds 4 WMMAs (one B frag + 4 accumulators live at a time -> no spills).
__global__ __launch_bounds__(256) void scores_kernel(const float* __restrict__ context,
                                                     const unsigned short* __restrict__ Wb,
                                                     const float* __restrict__ inp,
                                                     const float* __restrict__ b_ctx,
                                                     const float* __restrict__ Vv,
                                                     const unsigned char* __restrict__ mask,
                                                     float* __restrict__ scores) {
    __shared__ unsigned short As[SBLK * LDA];   // 65 KB bf16 context slab
    __shared__ float att_sh[SBLK];

    const int tid = threadIdx.x;
    const int b  = blockIdx.x >> 4;             // 16 s-blocks per batch row
    const int s0 = (blockIdx.x & 15) * SBLK;

    if (tid < SBLK) att_sh[tid] = 0.f;

    // Stage context[b, s0:s0+64, :] into LDS as bf16 (coalesced float4 loads)
    const float4* src = (const float4*)(context + (size_t)(b * Sdim + s0) * Ddim);
#pragma unroll
    for (int i = 0; i < (SBLK * Ddim / 4) / 256; ++i) {     // 32 iters
        int idx = tid + i * 256;
        int row = idx >> 7;                                  // Ddim/4 == 128
        int c4  = idx & 127;
        float4 v = src[idx];
        unsigned short* p = &As[row * LDA + c4 * 4];
        p[0] = f2bf(v.x); p[1] = f2bf(v.y); p[2] = f2bf(v.z); p[3] = f2bf(v.w);
    }
    __syncthreads();

    const int lane = tid & 31;
    const int wave = tid >> 5;
    const int l15  = lane & 15;
    const int sel  = lane >> 4;                 // which half-wave
    const float* inprow = inp + (size_t)b * Hdim;

    // A fragment row pointers for the 4 s-subtiles (M = lane%16 on both halves)
    const unsigned short* arow0 = &As[( 0 + l15) * LDA + sel * 8];
    const unsigned short* arow1 = &As[(16 + l15) * LDA + sel * 8];
    const unsigned short* arow2 = &As[(32 + l15) * LDA + sel * 8];
    const unsigned short* arow3 = &As[(48 + l15) * LDA + sel * 8];

    float sacc[4][8];
#pragma unroll
    for (int sb = 0; sb < 4; ++sb)
#pragma unroll
        for (int r = 0; r < 8; ++r) sacc[sb][r] = 0.f;

    for (int hti = 0; hti < 4; ++hti) {
        const int h = (wave + hti * 8) * 16 + l15;           // this lane's N column
        const unsigned short* wrow = Wb + (size_t)h * Ddim + sel * 8;

        v8f c0 = {}, c1 = {}, c2 = {}, c3 = {};
#pragma unroll 2
        for (int k0 = 0; k0 < Ddim; k0 += 32) {
            // B 32x16 bf16: lane<16 -> K {0..7,16..23}; lane>=16 -> {8..15,24..31}
            v8bf blo = *(const v8bf*)(wrow + k0);
            v8bf bhi = *(const v8bf*)(wrow + k0 + 16);
            v16bf bm = __builtin_shufflevector(blo, bhi, 0,1,2,3,4,5,6,7,8,9,10,11,12,13,14,15);

            // A 16x32 bf16: lane<16 -> K {0..7,16..23}; lane>=16 -> {8..15,24..31}
            {
                v8bf alo = *(const v8bf*)(arow0 + k0);
                v8bf ahi = *(const v8bf*)(arow0 + k0 + 16);
                v16bf a  = __builtin_shufflevector(alo, ahi, 0,1,2,3,4,5,6,7,8,9,10,11,12,13,14,15);
                c0 = __builtin_amdgcn_wmma_f32_16x16x32_bf16(false, a, false, bm, (short)0, c0, false, false);
            }
            {
                v8bf alo = *(const v8bf*)(arow1 + k0);
                v8bf ahi = *(const v8bf*)(arow1 + k0 + 16);
                v16bf a  = __builtin_shufflevector(alo, ahi, 0,1,2,3,4,5,6,7,8,9,10,11,12,13,14,15);
                c1 = __builtin_amdgcn_wmma_f32_16x16x32_bf16(false, a, false, bm, (short)0, c1, false, false);
            }
            {
                v8bf alo = *(const v8bf*)(arow2 + k0);
                v8bf ahi = *(const v8bf*)(arow2 + k0 + 16);
                v16bf a  = __builtin_shufflevector(alo, ahi, 0,1,2,3,4,5,6,7,8,9,10,11,12,13,14,15);
                c2 = __builtin_amdgcn_wmma_f32_16x16x32_bf16(false, a, false, bm, (short)0, c2, false, false);
            }
            {
                v8bf alo = *(const v8bf*)(arow3 + k0);
                v8bf ahi = *(const v8bf*)(arow3 + k0 + 16);
                v16bf a  = __builtin_shufflevector(alo, ahi, 0,1,2,3,4,5,6,7,8,9,10,11,12,13,14,15);
                c3 = __builtin_amdgcn_wmma_f32_16x16x32_bf16(false, a, false, bm, (short)0, c3, false, false);
            }
        }

        // ctx tile element (m = r + 8*sel, n = l15); add inp + b_ctx, tanh, weight by V
        const float vh = Vv[h];
        const float ih = inprow[h] + b_ctx[h];
#pragma unroll
        for (int r = 0; r < 8; ++r) {
            sacc[0][r] += vh * fast_tanhf(c0[r] + ih);
            sacc[1][r] += vh * fast_tanhf(c1[r] + ih);
            sacc[2][r] += vh * fast_tanhf(c2[r] + ih);
            sacc[3][r] += vh * fast_tanhf(c3[r] + ih);
        }
    }

    // reduce over N (16 lanes per half-wave), accumulate per-s into LDS
#pragma unroll
    for (int sb = 0; sb < 4; ++sb) {
#pragma unroll
        for (int r = 0; r < 8; ++r) {
            float x = sacc[sb][r];
            x += __shfl_xor(x, 1, 32);
            x += __shfl_xor(x, 2, 32);
            x += __shfl_xor(x, 4, 32);
            x += __shfl_xor(x, 8, 32);
            if (l15 == 0) atomicAdd(&att_sh[sb * 16 + sel * 8 + r], x);
        }
    }
    __syncthreads();

    if (tid < SBLK) {
        const size_t o = (size_t)b * Sdim + s0 + tid;
        scores[o] = mask[o] ? -__builtin_inff() : att_sh[tid];
    }
}

// ---------------- K3: row softmax over S=1024, in place ----------------
__global__ __launch_bounds__(256) void softmax_kernel(float* __restrict__ att) {
    __shared__ float redm[8];
    __shared__ float reds[8];
    const int tid = threadIdx.x, lane = tid & 31, wave = tid >> 5;
    float* row = att + (size_t)blockIdx.x * Sdim;

    float x[4];
#pragma unroll
    for (int j = 0; j < 4; ++j) x[j] = row[tid + j * 256];

    float m = fmaxf(fmaxf(x[0], x[1]), fmaxf(x[2], x[3]));
    for (int off = 16; off >= 1; off >>= 1) m = fmaxf(m, __shfl_xor(m, off, 32));
    if (lane == 0) redm[wave] = m;
    __syncthreads();
    float mm = redm[0];
#pragma unroll
    for (int w = 1; w < 8; ++w) mm = fmaxf(mm, redm[w]);

    float e[4], s = 0.f;
#pragma unroll
    for (int j = 0; j < 4; ++j) { e[j] = __expf(x[j] - mm); s += e[j]; }
    for (int off = 16; off >= 1; off >>= 1) s += __shfl_xor(s, off, 32);
    if (lane == 0) reds[wave] = s;
    __syncthreads();
    float tot = 0.f;
#pragma unroll
    for (int w = 0; w < 8; ++w) tot += reds[w];

    const float inv = 1.f / tot;
#pragma unroll
    for (int j = 0; j < 4; ++j) row[tid + j * 256] = e[j] * inv;
}

// ---------------- K4: cbar[b,d] = sum_s alpha[b,s] * context[b,s,d] ----------------
__global__ __launch_bounds__(512) void cbar_kernel(const float* __restrict__ context,
                                                   const float* __restrict__ alpha,
                                                   float* __restrict__ cbar) {
    __shared__ float al[Sdim];
    const int b = blockIdx.x, tid = threadIdx.x;
    for (int i = tid; i < Sdim; i += 512) al[i] = alpha[(size_t)b * Sdim + i];
    __syncthreads();
    const float* ctx = context + (size_t)b * Sdim * Ddim;
    float acc = 0.f;
#pragma unroll 4
    for (int s = 0; s < Sdim; ++s) acc = fmaf(al[s], ctx[(size_t)s * Ddim + tid], acc);
    cbar[(size_t)b * Ddim + tid] = acc;
}

extern "C" void kernel_launch(void* const* d_in, const int* in_sizes, int n_in,
                              void* d_out, int out_size, void* d_ws, size_t ws_size,
                              hipStream_t stream) {
    const float*         input   = (const float*)d_in[0];
    const float*         context = (const float*)d_in[1];
    const unsigned char* mask    = (const unsigned char*)d_in[2];   // jax bool_ = 1 byte
    const float*         W_in    = (const float*)d_in[3];
    const float*         b_in    = (const float*)d_in[4];
    const float*         W_ctx   = (const float*)d_in[5];
    const float*         b_ctx   = (const float*)d_in[6];
    const float*         V       = (const float*)d_in[7];

    float* hidden = (float*)d_out;                 // [B,H]
    float* alpha  = hidden + (size_t)Bdim * Hdim;  // [B,S] (raw scores first, then softmax'd)

    // workspace: inp [B,H] f32 | W_ctx bf16 [H,D] | cbar [B,D] f32  == 1 MB total
    float*          inp  = (float*)d_ws;
    unsigned short* Wb   = (unsigned short*)((char*)d_ws + (size_t)Bdim * Hdim * 4);
    float*          cbar = (float*)((char*)d_ws + (size_t)Bdim * Hdim * 4 + (size_t)Hdim * Ddim * 2);

    wcvt_kernel   <<<256, 256, 0, stream>>>(W_ctx, Wb, Hdim * Ddim);
    rowgemm_kernel<<<(Bdim * Hdim) / 256, 256, 0, stream>>>(input, W_in, b_in, inp);
    scores_kernel <<<Bdim * (Sdim / SBLK), 256, 0, stream>>>(context, Wb, inp, b_ctx, V, mask, alpha);
    softmax_kernel<<<Bdim, 256, 0, stream>>>(alpha);
    cbar_kernel   <<<Bdim, 512, 0, stream>>>(context, alpha, cbar);
    rowgemm_kernel<<<(Bdim * Hdim) / 256, 256, 0, stream>>>(cbar, W_ctx, b_ctx, hidden);
}
